// TL_NN5_18373870092975
// MI455X (gfx1250) — matrix operand order repaired and checked by reading, not compiled, not run on hardware
//
#include <hip/hip_runtime.h>
#include <hip/hip_bf16.h>

typedef __attribute__((ext_vector_type(16))) _Float16 v16h;
typedef __attribute__((ext_vector_type(8)))  _Float16 v8h;
typedef __attribute__((ext_vector_type(8)))  float    v8f;

#define TT 1024
#define BB 64
#define SIG10 4.5397868702434395e-05f   // sigmoid(-10) in f32

// workspace layout (bytes)
#define WS_PARTIAL   0                  // f32[3*64]   per-branch per-b accumulators
#define WS_A2SM      768                // f32[3*1024]
#define WS_SUF       13056              // f32[3*1025] suffix sums of A1sm
#define WS_P         25360              // f16[3*2048] zero-padded A1sm (first 1024 = 0)
#define WS_G         37648              // f16[3*64*1024] sigmoid matrix (16B aligned)

// ---------------------------------------------------------------------------
// Kernel 1: per-branch softmaxes, suffix sums, padded f16 weights, G matrix
// ---------------------------------------------------------------------------
__global__ __launch_bounds__(256) void prep_kernel(
    const float* __restrict__ x1, const float* __restrict__ x2, const float* __restrict__ x3,
    const float* __restrict__ t1, const float* __restrict__ t2, const float* __restrict__ t3,
    const float* __restrict__ b1, const float* __restrict__ b2, const float* __restrict__ b3,
    const float* __restrict__ A1_1, const float* __restrict__ A1_2, const float* __restrict__ A1_3,
    const float* __restrict__ A2_1, const float* __restrict__ A2_2, const float* __restrict__ A2_3,
    unsigned char* __restrict__ ws)
{
    __shared__ float sv[TT];
    __shared__ float red[256];
    const int br  = blockIdx.x;
    const int tid = threadIdx.x;

    const float* x  = (br == 0) ? x1   : (br == 1) ? x2   : x3;
    const float* tp = (br == 0) ? t1   : (br == 1) ? t2   : t3;
    const float* bp = (br == 0) ? b1   : (br == 1) ? b2   : b3;
    const float* A1 = (br == 0) ? A1_1 : (br == 1) ? A1_2 : A1_3;
    const float* A2 = (br == 0) ? A2_1 : (br == 1) ? A2_2 : A2_3;

    float*     partial = (float*)(ws + WS_PARTIAL);
    float*     a2sm    = (float*)(ws + WS_A2SM) + br * TT;
    float*     suf     = (float*)(ws + WS_SUF)  + br * (TT + 1);
    _Float16*  P       = (_Float16*)(ws + WS_P) + br * 2 * TT;
    _Float16*  g       = (_Float16*)(ws + WS_G) + br * BB * TT;

    if (tid < BB) partial[br * BB + tid] = 0.0f;   // fresh accumulators each launch

    // ---------------- softmax(A1) -> P (padded f16) + sv (f32) -------------
    for (int j = tid; j < TT; j += 256) sv[j] = A1[j];
    __syncthreads();
    float m = -3.4e38f;
    for (int j = tid; j < TT; j += 256) m = fmaxf(m, sv[j]);
    red[tid] = m; __syncthreads();
    for (int s = 128; s > 0; s >>= 1) { if (tid < s) red[tid] = fmaxf(red[tid], red[tid + s]); __syncthreads(); }
    float mx = red[0]; __syncthreads();
    float lsum = 0.0f;
    for (int j = tid; j < TT; j += 256) { float e = __expf(sv[j] - mx); sv[j] = e; lsum += e; }
    red[tid] = lsum; __syncthreads();
    for (int s = 128; s > 0; s >>= 1) { if (tid < s) red[tid] += red[tid + s]; __syncthreads(); }
    float inv = 1.0f / red[0]; __syncthreads();
    for (int j = tid; j < TT; j += 256) {
        float a = sv[j] * inv;
        sv[j] = a;
        P[j]      = (_Float16)0.0f;   // leading zero pad: Toeplitz "k < i" region
        P[TT + j] = (_Float16)a;
    }
    __syncthreads();
    if (tid == 0) {                    // suffix sums for the tail term
        float acc = 0.0f;
        suf[TT] = 0.0f;
        for (int q = TT - 1; q >= 0; --q) { acc += sv[q]; suf[q] = acc; }
    }
    __syncthreads();

    // ---------------- softmax(A2) -> a2sm (f32) ----------------------------
    for (int j = tid; j < TT; j += 256) sv[j] = A2[j];
    __syncthreads();
    m = -3.4e38f;
    for (int j = tid; j < TT; j += 256) m = fmaxf(m, sv[j]);
    red[tid] = m; __syncthreads();
    for (int s = 128; s > 0; s >>= 1) { if (tid < s) red[tid] = fmaxf(red[tid], red[tid + s]); __syncthreads(); }
    mx = red[0]; __syncthreads();
    lsum = 0.0f;
    for (int j = tid; j < TT; j += 256) { float e = __expf(sv[j] - mx); sv[j] = e; lsum += e; }
    red[tid] = lsum; __syncthreads();
    for (int s = 128; s > 0; s >>= 1) { if (tid < s) red[tid] += red[tid + s]; __syncthreads(); }
    inv = 1.0f / red[0]; __syncthreads();
    for (int j = tid; j < TT; j += 256) a2sm[j] = sv[j] * inv;

    // ---------------- G matrix: g[b,k] = sigmoid(bs - x*ts) ---------------
    const float ts = tp[0], bs = bp[0];
    for (int idx = tid; idx < BB * TT; idx += 256) {
        float xv = x[idx];
        float r  = bs - xv * ts;
        if (r == bs) r = -10.0f;                 // exact reference semantics
        float s = 1.0f / (1.0f + __expf(-r));
        g[idx] = (_Float16)s;
    }
}

// ---------------------------------------------------------------------------
// Kernel 2: WMMA GEMM  (G[64x1024] @ Toeplitz[1024x1024]) + weighted reduce
//   grid = (32, 3), block = 256 (8 waves); each wave owns one 16x16 tile
// ---------------------------------------------------------------------------
__global__ __launch_bounds__(256) void wmma_corr_kernel(
    const float* __restrict__ ba1, const float* __restrict__ ba2, const float* __restrict__ ba3,
    unsigned char* __restrict__ ws)
{
    __shared__ _Float16 Plds[2 * TT];
    const int br  = blockIdx.y;
    const int tid = threadIdx.x;

    const _Float16* P    = (const _Float16*)(ws + WS_P) + br * 2 * TT;
    const _Float16* g    = (const _Float16*)(ws + WS_G) + br * BB * TT;
    const float*    a2sm = (const float*)(ws + WS_A2SM) + br * TT;
    const float*    suf  = (const float*)(ws + WS_SUF)  + br * (TT + 1);
    float*          partial = (float*)(ws + WS_PARTIAL) + br * BB;
    const float beta_a = ((br == 0) ? ba1 : (br == 1) ? ba2 : ba3)[0];

    for (int j = tid; j < 2 * TT; j += 256) Plds[j] = P[j];
    __syncthreads();

    const int wave = tid >> 5;
    const int lane = tid & 31;
    const int tile = blockIdx.x * 8 + wave;       // 256 tiles per branch
    const int b0 = (tile >> 6) << 4;              // batch-row block
    const int i0 = (tile & 63) << 4;              // output-column block
    const int hi = (lane >> 4) & 1;
    const int n  = lane & 15;

    const _Float16* ga = g + (b0 + n) * TT;       // A-matrix row for this lane

    v8f c = {};
    const int kk0 = i0 & ~31;                     // skip the all-zero triangle
    for (int kk = kk0; kk < TT; kk += 32) {
        // 16-bit A 16x32 layout: lanes0-15 K 0-7/16-23, lanes16-31 K 8-15/24-31
        v8h lo = *(const v8h*)(ga + kk + hi * 8);
        v8h hv = *(const v8h*)(ga + kk + 16 + hi * 8);
        v16h a;
        #pragma unroll
        for (int h = 0; h < 8; ++h) { a[h] = lo[h]; a[h + 8] = hv[h]; }

        // 16-bit B 32x16 layout: lane n holds K = kk + 8*hi + h (h<8), kk + 16 + 8*hi + (h-8) (h>=8)
        // B[k][i0+n] = A1sm[k - i0 - n]  (zero pad handles k < i0+n)
        const int d0 = kk + hi * 8 - i0 - n + TT;
        v16h bf;
        #pragma unroll
        for (int h = 0; h < 16; ++h) bf[h] = Plds[d0 + h + (h >= 8 ? 8 : 0)];

        c = __builtin_amdgcn_wmma_f32_16x16x32_f16(
                false, a, false, bf, (short)0, c, false, false);
    }

    // C layout: lane -> column i = i0 + n ; VGPR r -> row b0 + 8*hi + r
    const int   i    = i0 + n;
    const float tail = SIG10 * suf[TT - i];
    const float base = 1.0f - beta_a + tail;
    const float w2   = a2sm[i];
    #pragma unroll
    for (int r = 0; r < 8; ++r) {
        float wb1 = c[r] + base;
        float act = fminf(fmaxf(wb1, 0.0f), 1.0f);
        atomicAdd(&partial[b0 + hi * 8 + r], w2 * (1.0f - act));
    }
}

// ---------------------------------------------------------------------------
// Kernel 3: combine three branches
// ---------------------------------------------------------------------------
__global__ void final_kernel(
    const float* __restrict__ bb1, const float* __restrict__ bb2, const float* __restrict__ bb3,
    const float* __restrict__ A4,  const float* __restrict__ beta4,
    const unsigned char* __restrict__ ws, float* __restrict__ out)
{
    const int b = threadIdx.x;
    if (b >= BB) return;
    const float* partial = (const float*)(ws + WS_PARTIAL);
    const float bb[3] = { bb1[0], bb2[0], bb3[0] };

    float a0 = A4[0], a1 = A4[1], a2 = A4[2];
    float mx = fmaxf(a0, fmaxf(a1, a2));
    float e0 = __expf(a0 - mx), e1 = __expf(a1 - mx), e2 = __expf(a2 - mx);
    float isum = 1.0f / (e0 + e1 + e2);
    const float w[3] = { e0 * isum, e1 * isum, e2 * isum };

    float acc = beta4[0];
    #pragma unroll
    for (int br = 0; br < 3; ++br) {
        float wb2 = bb[br] - partial[br * BB + b];
        float rr  = fminf(fmaxf(wb2, 0.0f), 1.0f);
        acc -= w[br] * (1.0f - rr);
    }
    out[b] = fminf(fmaxf(acc, 0.0f), 1.0f);
}

// ---------------------------------------------------------------------------
extern "C" void kernel_launch(void* const* d_in, const int* in_sizes, int n_in,
                              void* d_out, int out_size, void* d_ws, size_t ws_size,
                              hipStream_t stream) {
    const float* x1    = (const float*)d_in[0];
    const float* x2    = (const float*)d_in[1];
    const float* x3    = (const float*)d_in[2];
    const float* t1    = (const float*)d_in[3];
    const float* b1    = (const float*)d_in[4];
    const float* A1_1  = (const float*)d_in[5];
    const float* A2_1  = (const float*)d_in[6];
    const float* ba1   = (const float*)d_in[7];
    const float* bbb1  = (const float*)d_in[8];
    const float* t2    = (const float*)d_in[9];
    const float* b2    = (const float*)d_in[10];
    const float* A1_2  = (const float*)d_in[11];
    const float* A2_2  = (const float*)d_in[12];
    const float* ba2   = (const float*)d_in[13];
    const float* bbb2  = (const float*)d_in[14];
    const float* t3    = (const float*)d_in[15];
    const float* b3    = (const float*)d_in[16];
    const float* A1_3  = (const float*)d_in[17];
    const float* A2_3  = (const float*)d_in[18];
    const float* ba3   = (const float*)d_in[19];
    const float* bbb3  = (const float*)d_in[20];
    const float* A4    = (const float*)d_in[21];
    const float* beta4 = (const float*)d_in[22];
    unsigned char* ws  = (unsigned char*)d_ws;

    prep_kernel<<<3, 256, 0, stream>>>(x1, x2, x3, t1, t2, t3, b1, b2, b3,
                                       A1_1, A1_2, A1_3, A2_1, A2_2, A2_3, ws);

    dim3 g2(32, 3);
    wmma_corr_kernel<<<g2, 256, 0, stream>>>(ba1, ba2, ba3, ws);

    final_kernel<<<1, 64, 0, stream>>>(bbb1, bbb2, bbb3, A4, beta4, ws, (float*)d_out);
}
